// CGCNNLayer_23063974379747
// MI455X (gfx1250) — matrix operand order repaired and checked by reading, not compiled, not run on hardware
//
#include <hip/hip_runtime.h>
#include <hip/hip_bf16.h>

// ---------------------------------------------------------------------------
// CGCNN conv layer for MI455X (gfx1250, wave32, WMMA bf16)
//   pass0: z = concat(atom[src],atom[dst],nbr) @ W  -> column sum/sumsq
//   pass1: recompute z, bn1, sigmoid*softplus gate, atomic scatter to atoms
//   then bn2 over atoms + final softplus
// All index math is 32-bit: E*64 = 102M elems (410 MB) < 2^31.
// ---------------------------------------------------------------------------

typedef __attribute__((ext_vector_type(16))) __bf16 v16bf;
typedef __attribute__((ext_vector_type(8)))  __bf16 v8bf;
typedef __attribute__((ext_vector_type(2)))  __bf16 v2bf;
typedef __attribute__((ext_vector_type(8)))  float  v8f;

#define EPB 64          // edges per block (4 subtiles of 16)
#define ST  4           // subtiles per block
#define SA_STRIDE 200   // sA row stride in bf16 (400B: 16B aligned, conflict-free)
#define BN1_SLOTS 64
#define BN2_SLOTS 32

// workspace layout (float offsets)
#define WS_BN1_PART  0                      // BN1_SLOTS * 256 (sum[128], sumsq[128])
#define WS_BN1_SCALE 16384                  // 128
#define WS_BN1_SHIFT 16512                  // 128
#define WS_BN2_PART  16640                  // BN2_SLOTS * 128 (sum[64], sumsq[64])
#define WS_BN2_SCALE 20736                  // 64
#define WS_BN2_SHIFT 20800                  // 64
#define WS_WT        20864                  // Wt bf16 [128][192] = 12288 floats
#define WS_AU        33280                  // atom_update: N*64 floats (256-aligned)

// ---------------------------------------------------------------------------
__global__ __launch_bounds__(256) void zero_kernel(float* p, int n) {
  int i = blockIdx.x * blockDim.x + threadIdx.x;
  int stride = gridDim.x * blockDim.x;
  for (; i < n; i += stride) p[i] = 0.0f;
}

// Transpose W[192][128] fp32 -> Wt[128][192] bf16 in workspace (L2-resident).
__global__ __launch_bounds__(256) void wprep_kernel(const float* __restrict__ W,
                                                    float* wsF) {
  int idx = blockIdx.x * 256 + threadIdx.x;           // 0..24575
  if (idx < 192 * 128) {
    int k = idx >> 7, n = idx & 127;
    __bf16* Wt = (__bf16*)(wsF + WS_WT);
    Wt[n * 192 + k] = (__bf16)W[idx];
  }
}

// ---------------------------------------------------------------------------
template<int PASS>
__global__ __launch_bounds__(256) void gemm_pass(
    const float* __restrict__ atom, const float* __restrict__ nbr,
    const int* __restrict__ esrc, const int* __restrict__ edst,
    float* __restrict__ wsF, float* __restrict__ au, int E)
{
  __shared__ int    sIdx[2 * EPB];            // src[64], dst[64]
  __shared__ __bf16 sA[EPB * SA_STRIDE];      // 25.6 KB, shared by all 8 waves
  __shared__ float  zbuf[EPB * 128];          // 32 KB (used by PASS 1)

  const int tid  = threadIdx.x;
  const int wave = tid >> 5;
  const int lane = tid & 31;
  const int half = lane >> 4;
  const int l16  = lane & 15;
  const int n    = wave * 16 + l16;           // output column owned by lane
  const int tileBase = blockIdx.x * EPB;
  const bool fullTile = (tileBase + EPB) <= E;   // uniform: true for all but last

  // ---- stage edge indices
  if (tid < EPB) {
    int e  = tileBase + tid;
    int ec = (e < E) ? e : (E - 1);
    sIdx[tid]       = esrc[ec];
    sIdx[EPB + tid] = edst[ec];
  }
  __syncthreads();

  // ---- gather + convert A tile once per block: 64 edges x 192 k (bf16)
  //      coalesced float2 reads, packed bf16x2 LDS writes
  for (int i = tid; i < EPB * 32; i += 256) {
    int e  = i >> 5;
    int kk = (i & 31) * 2;
    int ec = tileBase + e;
    if (!fullTile && ec >= E) ec = E - 1;
    float2 vs = *(const float2*)(atom + sIdx[e] * 64 + kk);
    float2 vd = *(const float2*)(atom + sIdx[EPB + e] * 64 + kk);
    float2 vn = *(const float2*)(nbr + ec * 64 + kk);
    v2bf t;
    t[0] = (__bf16)vs.x; t[1] = (__bf16)vs.y;
    *(v2bf*)(sA + e * SA_STRIDE + kk) = t;
    t[0] = (__bf16)vd.x; t[1] = (__bf16)vd.y;
    *(v2bf*)(sA + e * SA_STRIDE + 64 + kk) = t;
    t[0] = (__bf16)vn.x; t[1] = (__bf16)vn.y;
    *(v2bf*)(sA + e * SA_STRIDE + 128 + kk) = t;
  }

  // ---- B fragments: Wt bf16 [n][k], contiguous 16-element (32B) loads.
  // B k order here (contiguous, same as A): elem i <-> K = ks*32 + 16*half + i.
  const __bf16* wp = (const __bf16*)(wsF + WS_WT) + n * 192 + 16 * half;
  v16bf bfrag[6];
#pragma unroll
  for (int ks = 0; ks < 6; ++ks)
    bfrag[ks] = *(const v16bf*)(wp + ks * 32);

  float bn_sc = 0.f, bn_sh = 0.f;
  if (PASS == 1) { bn_sc = wsF[WS_BN1_SCALE + n]; bn_sh = wsF[WS_BN1_SHIFT + n]; }

  __syncthreads();

  float s_sum = 0.f, s_sq = 0.f;

#pragma unroll
  for (int et = 0; et < ST; ++et) {
    // A k order here (contiguous, same as B): elem i <-> K = kb+16*half+i
    const __bf16* ap = sA + (et * 16 + l16) * SA_STRIDE + 16 * half;

    v8f acc = {0.f, 0.f, 0.f, 0.f, 0.f, 0.f, 0.f, 0.f};
#pragma unroll
    for (int ks = 0; ks < 6; ++ks) {
      v8bf lo = *(const v8bf*)(ap + ks * 32);
      v8bf hi = *(const v8bf*)(ap + ks * 32 + 8);
      v16bf a = __builtin_shufflevector(lo, hi,
          0, 1, 2, 3, 4, 5, 6, 7, 8, 9, 10, 11, 12, 13, 14, 15);
      acc = __builtin_amdgcn_wmma_f32_16x16x32_bf16(
          false, a, false, bfrag[ks], (short)0, acc, false, false);
    }

    const int mrow0 = et * 16 + 8 * half;     // local row of acc[0]
    if (PASS == 0) {
      if (fullTile) {
#pragma unroll
        for (int r = 0; r < 8; ++r) {
          float z = acc[r];
          s_sum += z; s_sq += z * z;
        }
      } else {
#pragma unroll
        for (int r = 0; r < 8; ++r) {
          if (tileBase + mrow0 + r < E) {
            float z = acc[r];
            s_sum += z; s_sq += z * z;
          }
        }
      }
    } else {
#pragma unroll
      for (int r = 0; r < 8; ++r)
        zbuf[(mrow0 + r) * 128 + n] = acc[r] * bn_sc + bn_sh;
    }
  }

  if (PASS == 0) {
    // combine the two half-wave row groups, then spread-atomic the partials
    s_sum += __shfl_xor(s_sum, 16, 32);
    s_sq  += __shfl_xor(s_sq, 16, 32);
    if (half == 0) {
      int slot = blockIdx.x & (BN1_SLOTS - 1);
      atomicAdd(&wsF[WS_BN1_PART + slot * 256 + n],       s_sum);
      atomicAdd(&wsF[WS_BN1_PART + slot * 256 + 128 + n], s_sq);
    }
  } else {
    __syncthreads();
    // gate + scatter: msg[e][d] = sigmoid(z[d]) * softplus(z[64+d])
    if (fullTile) {
      for (int i = tid; i < EPB * 64; i += 256) {
        int el = i >> 6, d = i & 63;
        float f = zbuf[el * 128 + d];
        float c = zbuf[el * 128 + 64 + d];
        float sig = 1.0f / (1.0f + __expf(-f));
        float sp  = (c > 20.0f) ? c : log1pf(__expf(c));
        atomicAdd(au + sIdx[EPB + el] * 64 + d, sig * sp);
      }
    } else {
      for (int i = tid; i < EPB * 64; i += 256) {
        int el = i >> 6, d = i & 63;
        if (tileBase + el < E) {
          float f = zbuf[el * 128 + d];
          float c = zbuf[el * 128 + 64 + d];
          float sig = 1.0f / (1.0f + __expf(-f));
          float sp  = (c > 20.0f) ? c : log1pf(__expf(c));
          atomicAdd(au + sIdx[EPB + el] * 64 + d, sig * sp);
        }
      }
    }
  }
}

// ---------------------------------------------------------------------------
__global__ void bn1_finalize(const float* __restrict__ gamma,
                             const float* __restrict__ beta,
                             float* wsF, int E) {
  int nidx = threadIdx.x;                     // 128 threads
  float s = 0.f, s2 = 0.f;
  for (int slot = 0; slot < BN1_SLOTS; ++slot) {
    s  += wsF[WS_BN1_PART + slot * 256 + nidx];
    s2 += wsF[WS_BN1_PART + slot * 256 + 128 + nidx];
  }
  float mean = s / (float)E;
  float var  = s2 / (float)E - mean * mean;
  if (var < 0.f) var = 0.f;
  float sc = gamma[nidx] * rsqrtf(var + 1e-5f);
  wsF[WS_BN1_SCALE + nidx] = sc;
  wsF[WS_BN1_SHIFT + nidx] = beta[nidx] - mean * sc;
}

__global__ __launch_bounds__(256) void bn2_reduce(const float* __restrict__ au,
                                                  float* wsF, int N) {
  __shared__ float ls[128];
  int tid = threadIdx.x;
  if (tid < 128) ls[tid] = 0.f;
  __syncthreads();
  float s = 0.f, s2 = 0.f;
  int total = N * 64;
  int stride = gridDim.x * 256;               // multiple of 64 -> col fixed/thread
  for (int i = blockIdx.x * 256 + tid; i < total; i += stride) {
    float v = au[i];
    s += v; s2 += v * v;
  }
  int c = tid & 63;
  atomicAdd(&ls[c], s);
  atomicAdd(&ls[64 + c], s2);
  __syncthreads();
  if (tid < 128) {
    int slot = blockIdx.x & (BN2_SLOTS - 1);
    atomicAdd(&wsF[WS_BN2_PART + slot * 128 + tid], ls[tid]);
  }
}

__global__ void bn2_finalize(const float* __restrict__ gamma,
                             const float* __restrict__ beta,
                             float* wsF, int N) {
  int c = threadIdx.x;                        // 64 threads
  float s = 0.f, s2 = 0.f;
  for (int slot = 0; slot < BN2_SLOTS; ++slot) {
    s  += wsF[WS_BN2_PART + slot * 128 + c];
    s2 += wsF[WS_BN2_PART + slot * 128 + 64 + c];
  }
  float mean = s / (float)N;
  float var  = s2 / (float)N - mean * mean;
  if (var < 0.f) var = 0.f;
  float sc = gamma[c] * rsqrtf(var + 1e-5f);
  wsF[WS_BN2_SCALE + c] = sc;
  wsF[WS_BN2_SHIFT + c] = beta[c] - mean * sc;
}

__global__ __launch_bounds__(256) void final_kernel(
    const float* __restrict__ atom, const float* __restrict__ au,
    const float* __restrict__ wsF, float* __restrict__ out, int total) {
  int i = blockIdx.x * 256 + threadIdx.x;
  if (i < total) {
    int c = i & 63;
    float v = atom[i] + au[i] * wsF[WS_BN2_SCALE + c] + wsF[WS_BN2_SHIFT + c];
    out[i] = (v > 20.0f) ? v : log1pf(__expf(v));
  }
}

// ---------------------------------------------------------------------------
extern "C" void kernel_launch(void* const* d_in, const int* in_sizes, int n_in,
                              void* d_out, int out_size, void* d_ws, size_t ws_size,
                              hipStream_t stream) {
  const float* atom = (const float*)d_in[0];   // [N,64]
  const float* nbr  = (const float*)d_in[1];   // [E,64]
  const int*   esrc = (const int*)d_in[2];     // [E]
  const int*   edst = (const int*)d_in[3];     // [E]
  const float* W    = (const float*)d_in[4];   // [192,128]
  const float* bias = (const float*)d_in[5];   // [128] (cancelled by bn1)
  const float* g1   = (const float*)d_in[6];
  const float* b1   = (const float*)d_in[7];
  const float* g2   = (const float*)d_in[8];
  const float* b2   = (const float*)d_in[9];
  (void)bias; (void)n_in; (void)ws_size; (void)out_size;

  int N = in_sizes[0] / 64;
  int E = in_sizes[2];

  float* wsF = (float*)d_ws;
  float* au  = wsF + WS_AU;                    // atom_update accumulator [N,64]

  int zeroN = WS_AU + N * 64;                  // covers stats + Wt + accumulator
  zero_kernel<<<1024, 256, 0, stream>>>(wsF, zeroN);
  wprep_kernel<<<96, 256, 0, stream>>>(W, wsF);

  int nTiles = (E + EPB - 1) / EPB;
  gemm_pass<0><<<nTiles, 256, 0, stream>>>(atom, nbr, esrc, edst, wsF, au, E);
  bn1_finalize<<<1, 128, 0, stream>>>(g1, b1, wsF, E);
  gemm_pass<1><<<nTiles, 256, 0, stream>>>(atom, nbr, esrc, edst, wsF, au, E);
  bn2_reduce<<<512, 256, 0, stream>>>(au, wsF, N);
  bn2_finalize<<<1, 64, 0, stream>>>(g2, b2, wsF, N);

  int total = N * 64;
  final_kernel<<<(total + 255) / 256, 256, 0, stream>>>(
      atom, au, wsF, (float*)d_out, total);
}
